// MAC_Cell_70214125355595
// MI455X (gfx1250) — matrix operand order repaired and checked by reading, not compiled, not run on hardware
//
#include <hip/hip_runtime.h>
#include <hip/hip_bf16.h>

// ---------------------------------------------------------------------------
// MAC cell forward, algebraically collapsed:
//   retrieve[b,k] = knowledge[b,k,:] . t_b + const_b
// with t_b = r_Wd_top^T ncr_b + r_Wk (mem_proj_b * (r_Wd_bot^T ncr_b)),
// ncr = new_control * r_Wr.  All small dense GEMMs run on
// v_wmma_f32_16x16x32_bf16; knowledge is streamed twice (memory-bound,
// ~270 MB -> ~12us at 23.3 TB/s).
// ---------------------------------------------------------------------------

typedef __attribute__((ext_vector_type(16))) __bf16 v16bf;
typedef __attribute__((ext_vector_type(8)))  float  v8f;

#define DIMD 512
#define BATCH 128
#define KNOW 512
#define SQLEN 128
#define NSTEP 12

union Pk16 {            // 16 bf16 = 32 bytes
  __bf16 h[16];
  uint4  q[2];
  v16bf  v;
};
union Pk4 {             // 4 bf16 = 8 bytes
  __bf16 h[4];
  uint2  q;
};

__device__ __forceinline__ float wave_reduce_sum(float v) {
#pragma unroll
  for (int off = 16; off > 0; off >>= 1) v += __shfl_xor(v, off, 32);
  return v;
}
__device__ __forceinline__ float wave_reduce_max(float v) {
#pragma unroll
  for (int off = 16; off > 0; off >>= 1) v = fmaxf(v, __shfl_xor(v, off, 32));
  return v;
}

// dot of two 512-float vectors across one wave32 (lane L owns floats [16L,16L+16))
__device__ __forceinline__ float wave_dot512(const float* __restrict__ a,
                                             const float* __restrict__ b) {
  int l = threadIdx.x & 31;
  float s = 0.f;
#pragma unroll
  for (int i = 0; i < 4; ++i) {
    int idx = l * 16 + i * 4;
    float4 va = *(const float4*)(a + idx);
    float4 vb = *(const float4*)(b + idx);
    s += va.x * vb.x + va.y * vb.y + va.z * vb.z + va.w * vb.w;
  }
  return wave_reduce_sum(s);
}

// ---------------------------------------------------------------------------
// Generic bf16 WMMA GEMM:  C[M,N] = segA1 + segA2 (+bias +bias2 +Cin)
//   segment: A (M x K, fp32, row stride lda) x B (K x N, or N x K if TRANSB)
//   Amul (HASMUL): elementwise multiplier on A1, same indexing as A1.
// Block = 128 threads (4 waves); block tile = 16(M) x 128(N); K chunk = 64.
// Each wave owns a 16x32 slice -> 4 WMMAs per staging round.
// B is staged in FRAGMENT-MAJOR LDS layout: for (chunk c, subtile s, half g,
// col n) the lane's 16 bf16 values are contiguous -> B frag = 2x ds_load_b128.
// A tile is k-contiguous per row -> A frag = 2x ds_load_b128.
// Grid = (M/16, N/128).
// ---------------------------------------------------------------------------
template <bool TRANSB, bool HASMUL>
__global__ __launch_bounds__(128) void wmma_gemm_t(
    const float* __restrict__ A1, int lda1, const float* __restrict__ Amul,
    const float* __restrict__ B1, int ldb1, int K1,
    const float* __restrict__ A2, int lda2,
    const float* __restrict__ B2, int ldb2, int K2,
    const float* __restrict__ bias, const float* __restrict__ bias2,
    const float* __restrict__ Cin, float* __restrict__ C, int ldc) {
  __shared__ __align__(16) __bf16 lAs[16 * 64];   // A tile [m][k]      (2 KB)
  // B frag-major: index = ((((c*8 + s)*2 + g)*16 + n) * 16 + i)        (16 KB)
  __shared__ __align__(16) __bf16 lBf[2 * 8 * 2 * 16 * 16];

  const int tid  = threadIdx.x;
  const int m0   = blockIdx.x * 16;
  const int nblk = blockIdx.y * 128;
  const int w    = tid >> 5;
  const int lane = tid & 31;
  const int g    = lane >> 4;   // lane half
  const int n    = lane & 15;   // N within 16-tile (B/C/D); M for A frag
  const int nw   = w * 32;      // wave's N slice within block tile

  v8f acc0 = {}, acc1 = {};

  for (int seg = 0; seg < 2; ++seg) {
    const float* A   = seg ? A2 : A1;
    const float* Bm  = seg ? B2 : B1;
    const int    K   = seg ? K2 : K1;
    const int    lda = seg ? lda2 : lda1;
    const int    ldb = seg ? ldb2 : ldb1;
    if (A == nullptr || K == 0) continue;

    for (int kk = 0; kk < K; kk += 64) {
      __syncthreads();  // protect LDS from previous round's readers
      // ---- stage A 16x64 (256 float4 / 128 threads = 2 each) ----
#pragma unroll
      for (int i = 0; i < 2; ++i) {
        int vi = tid + i * 128;
        int r = vi >> 4, c4 = vi & 15;
        const float* ap = A + (long)(m0 + r) * lda + kk + c4 * 4;
        float4 v = *(const float4*)ap;
        if (HASMUL) {
          float4 mv = *(const float4*)(Amul + (long)(m0 + r) * lda + kk + c4 * 4);
          v.x *= mv.x; v.y *= mv.y; v.z *= mv.z; v.w *= mv.w;
        }
        Pk4 pk;
        pk.h[0] = (__bf16)v.x; pk.h[1] = (__bf16)v.y;
        pk.h[2] = (__bf16)v.z; pk.h[3] = (__bf16)v.w;
        *(uint2*)(lAs + r * 64 + c4 * 4) = pk.q;
      }
      // ---- stage B 64x128 into frag-major layout ----
      // 512 lane-rows of 16 k-values; thread handles lr = tid + j*128.
      // lr = c*256 + s*32 + g*16 + n  (matches frag-major index / 16)
#pragma unroll
      for (int j = 0; j < 4; ++j) {
        int lr = tid + j * 128;
        int n_ = lr & 15, g_ = (lr >> 4) & 1, s_ = (lr >> 5) & 7, c_ = lr >> 8;
        int kbase = kk + c_ * 32 + g_ * 16;
        Pk16 pk;
        if (TRANSB) {
          // global row (nblk + s*16 + n), 16 consecutive k: 4x float4
          const float* bp = Bm + (long)(nblk + s_ * 16 + n_) * ldb + kbase;
#pragma unroll
          for (int q = 0; q < 4; ++q) {
            float4 v = *(const float4*)(bp + q * 4);
            pk.h[q * 4 + 0] = (__bf16)v.x; pk.h[q * 4 + 1] = (__bf16)v.y;
            pk.h[q * 4 + 2] = (__bf16)v.z; pk.h[q * 4 + 3] = (__bf16)v.w;
          }
        } else {
          // global column (nblk + s*16 + n): 16 b32 loads, coalesced across
          // threads (consecutive threads -> consecutive columns)
          int col = nblk + s_ * 16 + n_;
#pragma unroll
          for (int i = 0; i < 16; ++i)
            pk.h[i] = (__bf16)Bm[(long)(kbase + i) * ldb + col];
        }
        *(uint4*)(lBf + lr * 16 + 0) = pk.q[0];
        *(uint4*)(lBf + lr * 16 + 8) = pk.q[1];
      }
      __syncthreads();

      // ---- 2 K-chunks x 2 N-subtiles = 4 WMMAs per round per wave ----
#pragma unroll
      for (int c = 0; c < 2; ++c) {
        // A frag (16-bit A 16x32, contiguous k as in B): lane&15 = M;
        // VGPR0-7: K = c*32 + g*16 + [0..15]
        Pk16 a;
        a.q[0] = *(const uint4*)(lAs + n * 64 + c * 32 + g * 16);
        a.q[1] = *(const uint4*)(lAs + n * 64 + c * 32 + g * 16 + 8);
        // B frags: contiguous 32B per lane in frag-major layout
        const v16bf b0 =
            *(const v16bf*)(lBf + ((((c * 8 + (w * 2 + 0)) * 2 + g) * 16 + n) << 4));
        const v16bf b1 =
            *(const v16bf*)(lBf + ((((c * 8 + (w * 2 + 1)) * 2 + g) * 16 + n) << 4));
        acc0 = __builtin_amdgcn_wmma_f32_16x16x32_bf16(
            false, a.v, false, b0, (short)0, acc0, false, false);
        acc1 = __builtin_amdgcn_wmma_f32_16x16x32_bf16(
            false, a.v, false, b1, (short)0, acc1, false, false);
      }
    }
  }

  // C/D layout: lane group g, VGPR r -> row m0 + g*8 + r, col base + n
#pragma unroll
  for (int r = 0; r < 8; ++r) {
    int row  = m0 + g * 8 + r;
    int col0 = nblk + nw + n;
    int col1 = col0 + 16;
    float v0 = acc0[r], v1 = acc1[r];
    if (bias)  { v0 += bias[col0];  v1 += bias[col1]; }
    if (bias2) { v0 += bias2[col0]; v1 += bias2[col1]; }
    if (Cin) {
      v0 += Cin[(long)row * ldc + col0];
      v1 += Cin[(long)row * ldc + col1];
    }
    C[(long)row * ldc + col0] = v0;
    C[(long)row * ldc + col1] = v1;
  }
}

// ---------------------------------------------------------------------------
// Question attention: focus logits, softmax over SQ, new_control (mean),
// and ncr = new_control * r_Wr.  One block per batch row.
// ---------------------------------------------------------------------------
__global__ __launch_bounds__(256) void control_kernel(
    const float* __restrict__ cq, const float* __restrict__ Wf,
    const float* __restrict__ bfv, const float* __restrict__ question,
    const float* __restrict__ Wr, float* __restrict__ nc,
    float* __restrict__ ncr) {
  int b = blockIdx.x;
  int t = threadIdx.x, wave = t >> 5, lane = t & 31;
  __shared__ __align__(16) float cqw[DIMD];
  __shared__ float p[SQLEN];
  __shared__ float red2[SQLEN];

  for (int i = t; i < DIMD; i += 256) cqw[i] = cq[b * DIMD + i] * Wf[i];
  __syncthreads();

  const float* qb = question + (long)b * SQLEN * DIMD;
  for (int s = wave; s < SQLEN; s += 8) {
    if (s + 8 < SQLEN)
      __builtin_prefetch(qb + (long)(s + 8) * DIMD + lane * 16, 0, 1);
    float d = wave_dot512(qb + (long)s * DIMD, cqw);
    if (lane == 0) p[s] = d + bfv[0];
  }
  __syncthreads();

  if (t < SQLEN) red2[t] = p[t];
  __syncthreads();
  for (int off = 64; off > 0; off >>= 1) {
    if (t < off) red2[t] = fmaxf(red2[t], red2[t + off]);
    __syncthreads();
  }
  float m = red2[0];
  __syncthreads();
  if (t < SQLEN) { p[t] = __expf(p[t] - m); red2[t] = p[t]; }
  __syncthreads();
  for (int off = 64; off > 0; off >>= 1) {
    if (t < off) red2[t] += red2[t + off];
    __syncthreads();
  }
  float inv = 1.f / (red2[0] * (float)SQLEN);  // softmax * (1/SQ) mean
  __syncthreads();

  for (int d = t; d < DIMD; d += 256) {
    float acc = 0.f;
    for (int s = 0; s < SQLEN; ++s) acc += p[s] * qb[(long)s * DIMD + d];
    float v = acc * inv;
    nc[b * DIMD + d]  = v;
    ncr[b * DIMD + d] = v * Wr[d];
  }
}

// w = mem_proj * V ; const_b = r_bk.w + r_bd.ncr + r_br
__global__ __launch_bounds__(256) void wconst_kernel(
    const float* __restrict__ memproj, const float* __restrict__ V,
    const float* __restrict__ ncr, const float* __restrict__ bk,
    const float* __restrict__ bd, const float* __restrict__ br,
    float* __restrict__ w, float* __restrict__ constv) {
  int b = blockIdx.x, t = threadIdx.x;
  float acc = 0.f;
  for (int d = t; d < DIMD; d += 256) {
    float wv = memproj[b * DIMD + d] * V[b * DIMD + d];
    w[b * DIMD + d] = wv;
    acc += bk[d] * wv + bd[d] * ncr[b * DIMD + d];
  }
  acc = wave_reduce_sum(acc);
  __shared__ float red[8];
  if ((t & 31) == 0) red[t >> 5] = acc;
  __syncthreads();
  if (t == 0) {
    float s = 0.f;
#pragma unroll
    for (int i = 0; i < 8; ++i) s += red[i];
    constv[b] = s + br[0];
  }
}

// retrieve logits: knowledge[b,k,:] . t_b + const_b   (streaming pass 1)
__global__ __launch_bounds__(256) void retrieve_kernel(
    const float* __restrict__ knowledge, const float* __restrict__ tvec,
    const float* __restrict__ constv, float* __restrict__ logits) {
  int b = blockIdx.x, slab = blockIdx.y;  // 8 slabs x 64 rows
  int wave = threadIdx.x >> 5, lane = threadIdx.x & 31;
  __shared__ __align__(16) float ts[DIMD];
  for (int i = threadIdx.x; i < DIMD; i += 256) ts[i] = tvec[b * DIMD + i];
  __syncthreads();
  const float* kb = knowledge + ((long)b * KNOW + slab * 64) * DIMD;
  float cb = constv[b];
  for (int r = wave; r < 64; r += 8) {
    if (r + 8 < 64)
      __builtin_prefetch(kb + (long)(r + 8) * DIMD + lane * 16, 0, 1);
    float s = wave_dot512(kb + (long)r * DIMD, ts);
    if (lane == 0) logits[b * KNOW + slab * 64 + r] = s + cb;
  }
}

// softmax over K=512 per batch row
__global__ __launch_bounds__(256) void softmax512_kernel(
    const float* __restrict__ logits, float* __restrict__ attn) {
  int b = blockIdx.x, t = threadIdx.x;
  __shared__ float red[8];
  float v0 = logits[b * KNOW + t];
  float v1 = logits[b * KNOW + 256 + t];
  float m = wave_reduce_max(fmaxf(v0, v1));
  if ((t & 31) == 0) red[t >> 5] = m;
  __syncthreads();
  float mm = red[0];
#pragma unroll
  for (int i = 1; i < 8; ++i) mm = fmaxf(mm, red[i]);
  __syncthreads();
  float e0 = __expf(v0 - mm), e1 = __expf(v1 - mm);
  float s = wave_reduce_sum(e0 + e1);
  if ((t & 31) == 0) red[t >> 5] = s;
  __syncthreads();
  float ss = 0.f;
#pragma unroll
  for (int i = 0; i < 8; ++i) ss += red[i];
  float inv = 1.f / ss;
  attn[b * KNOW + t]       = e0 * inv;
  attn[b * KNOW + 256 + t] = e1 * inv;
}

// read[b,:] = (1/K) * attn[b,:] @ knowledge[b]   (streaming pass 2)
__global__ __launch_bounds__(256) void read_kernel(
    const float* __restrict__ knowledge, const float* __restrict__ attn,
    float* __restrict__ readv) {
  int b = blockIdx.x, t = threadIdx.x;
  __shared__ float p[KNOW];
  for (int i = t; i < KNOW; i += 256) p[i] = attn[b * KNOW + i];
  __syncthreads();
  const float* kb = knowledge + (long)b * KNOW * DIMD;
  float a0 = 0.f, a1 = 0.f;
  for (int k = 0; k < KNOW; ++k) {
    if ((k & 7) == 0 && k + 8 < KNOW)
      __builtin_prefetch(kb + (long)(k + 8) * DIMD + t, 0, 1);
    float pk = p[k];
    a0 += pk * kb[(long)k * DIMD + t];
    a1 += pk * kb[(long)k * DIMD + t + 256];
  }
  readv[b * DIMD + t]       = a0 * (1.f / (float)KNOW);
  readv[b * DIMD + t + 256] = a1 * (1.f / (float)KNOW);
}

// step attention over h_controls/h_memories (with exact-zero pad mask) + gate
__global__ __launch_bounds__(256) void writeattn_kernel(
    const float* __restrict__ h_controls, const float* __restrict__ h_memories,
    const float* __restrict__ nc, const float* __restrict__ Wca,
    const float* __restrict__ bca, const float* __restrict__ Wm3,
    const float* __restrict__ bm3, float* __restrict__ msa,
    float* __restrict__ gate) {
  int b = blockIdx.x, t = threadIdx.x, wave = t >> 5, lane = t & 31;
  __shared__ __align__(16) float ncw[DIMD];
  __shared__ float ca[NSTEP], p[NSTEP], gred[8];

  float gacc = 0.f;
  for (int i = t; i < DIMD; i += 256) {
    float v = nc[b * DIMD + i];
    ncw[i] = v * Wca[i];
    gacc += v * Wm3[i];
  }
  __syncthreads();

  const float* hc = h_controls + (long)b * NSTEP * DIMD;
  const float* hm = h_memories + (long)b * NSTEP * DIMD;
  for (int s = wave; s < NSTEP; s += 8) {
    float d = wave_dot512(hc + (long)s * DIMD, ncw);
    if (lane == 0) ca[s] = d + bca[0];
  }
  gacc = wave_reduce_sum(gacc);
  if ((t & 31) == 0) gred[t >> 5] = gacc;
  __syncthreads();

  if (t == 0) {
    float gs = 0.f;
#pragma unroll
    for (int i = 0; i < 8; ++i) gs += gred[i];
    gate[b] = 1.f / (1.f + __expf(-(gs + bm3[0])));
    float m = -1e30f;
    for (int s = 0; s < NSTEP; ++s) {
      float v = ca[s];
      if (v == 0.f) v += -1000000000.0f;  // pad mask
      ca[s] = v;
      m = fmaxf(m, v);
    }
    float sum = 0.f;
    for (int s = 0; s < NSTEP; ++s) { p[s] = __expf(ca[s] - m); sum += p[s]; }
    float inv = 1.f / sum;
    for (int s = 0; s < NSTEP; ++s) p[s] *= inv;
  }
  __syncthreads();

  float a0 = 0.f, a1 = 0.f;
  for (int s = 0; s < NSTEP; ++s) {
    a0 += p[s] * hm[(long)s * DIMD + t];
    a1 += p[s] * hm[(long)s * DIMD + t + 256];
  }
  msa[b * DIMD + t]       = a0;
  msa[b * DIMD + t + 256] = a1;
}

// final gating + output assembly: out[0]=all_controls, out[1]=all_memories
__global__ __launch_bounds__(256) void output_kernel(
    const float* __restrict__ nc, const float* __restrict__ mp,
    const float* __restrict__ gate, const float* __restrict__ h_controls,
    const float* __restrict__ h_memories, float* __restrict__ out) {
  long i = (long)blockIdx.x * blockDim.x + threadIdx.x;
  const long TOT = (long)BATCH * NSTEP * DIMD;
  if (i >= TOT) return;
  int d = (int)(i & (DIMD - 1));
  long bs = i >> 9;
  int s = (int)(bs % NSTEP);
  int b = (int)(bs / NSTEP);
  float c, m;
  if (s == 0) {
    c = nc[b * DIMD + d];
    float g = gate[b];
    m = mp[b * DIMD + d] * g + (1.f - g) * h_memories[(long)b * NSTEP * DIMD + d];
  } else {
    c = h_controls[((long)b * NSTEP + s - 1) * DIMD + d];
    m = h_memories[((long)b * NSTEP + s - 1) * DIMD + d];
  }
  out[i]       = c;
  out[TOT + i] = m;
}

extern "C" void kernel_launch(void* const* d_in, const int* in_sizes, int n_in,
                              void* d_out, int out_size, void* d_ws,
                              size_t ws_size, hipStream_t stream) {
  const float* x            = (const float*)d_in[0];
  const float* h_controls   = (const float*)d_in[1];
  const float* h_memories   = (const float*)d_in[2];
  const float* knowledge    = (const float*)d_in[3];
  const float* question     = (const float*)d_in[4];
  const float* question_rep = (const float*)d_in[5];
  const float* W_qs  = (const float*)d_in[6];  const float* b_qs  = (const float*)d_in[7];
  const float* c_Wcq = (const float*)d_in[8];  const float* c_bcq = (const float*)d_in[9];
  const float* c_Wf  = (const float*)d_in[10]; const float* c_bf  = (const float*)d_in[11];
  const float* r_Wm  = (const float*)d_in[12]; const float* r_bm  = (const float*)d_in[13];
  const float* r_Wk  = (const float*)d_in[14]; const float* r_bk  = (const float*)d_in[15];
  const float* r_Wd  = (const float*)d_in[16]; const float* r_bd  = (const float*)d_in[17];
  const float* r_Wr  = (const float*)d_in[18]; const float* r_br  = (const float*)d_in[19];
  const float* w_Wm1 = (const float*)d_in[20]; const float* w_bm1 = (const float*)d_in[21];
  const float* w_Wca = (const float*)d_in[22]; const float* w_bca = (const float*)d_in[23];
  const float* w_Wm2 = (const float*)d_in[24]; const float* w_bm2 = (const float*)d_in[25];
  const float* w_Ws  = (const float*)d_in[26]; const float* w_bs  = (const float*)d_in[27];
  const float* w_Wm3 = (const float*)d_in[28]; const float* w_bm3 = (const float*)d_in[29];

  float* out = (float*)d_out;
  float* ws  = (float*)d_ws;
  const long BD = (long)BATCH * DIMD;  // 65536 floats per slab

  float* quest   = ws + 0 * BD;
  float* memproj = ws + 1 * BD;
  float* cq      = ws + 2 * BD;
  float* nc      = ws + 3 * BD;
  float* ncr     = ws + 4 * BD;
  float* U       = ws + 5 * BD;
  float* V       = ws + 6 * BD;
  float* wv      = ws + 7 * BD;
  float* tvec    = ws + 8 * BD;
  float* logits  = ws + 9 * BD;   // B*K
  float* rattn   = ws + 10 * BD;  // B*K
  float* readv   = ws + 11 * BD;
  float* m1      = ws + 12 * BD;
  float* msa     = ws + 13 * BD;
  float* mpv     = ws + 14 * BD;
  float* constv  = ws + 15 * BD;
  float* gate    = constv + BATCH;

  dim3 ggrid(BATCH / 16, DIMD / 128);  // (8, 4)
  dim3 gblk(128);
  const int HLD = NSTEP * DIMD;        // 6144: row stride of h_controls/h_memories
  const int DD  = DIMD * DIMD;         // 262144: bottom half offset of 2D weights

  // 1) quest_state = (x * question_rep) @ W_qs + b_qs
  wmma_gemm_t<false, true><<<ggrid, gblk, 0, stream>>>(
      x, DIMD, question_rep, W_qs, DIMD, DIMD,
      nullptr, 0, nullptr, 0, 0, b_qs, nullptr, nullptr, quest, DIMD);
  // 2) mem_proj = prev_memory @ r_Wm + r_bm   (prev_memory = h_memories[:,0])
  wmma_gemm_t<false, false><<<ggrid, gblk, 0, stream>>>(
      h_memories, HLD, nullptr, r_Wm, DIMD, DIMD,
      nullptr, 0, nullptr, 0, 0, r_bm, nullptr, nullptr, memproj, DIMD);
  // 3) cq = [quest_state, prev_control] @ c_Wcq + c_bcq
  wmma_gemm_t<false, false><<<ggrid, gblk, 0, stream>>>(
      quest, DIMD, nullptr, c_Wcq, DIMD, DIMD,
      h_controls, HLD, c_Wcq + DD, DIMD, DIMD, c_bcq, nullptr, nullptr, cq, DIMD);
  // 4) question attention -> new_control, ncr
  control_kernel<<<BATCH, 256, 0, stream>>>(cq, c_Wf, c_bf, question, r_Wr, nc, ncr);
  // 5,6) U = ncr @ r_Wd_top^T ; V = ncr @ r_Wd_bot^T   (transB)
  wmma_gemm_t<true, false><<<ggrid, gblk, 0, stream>>>(
      ncr, DIMD, nullptr, r_Wd, DIMD, DIMD,
      nullptr, 0, nullptr, 0, 0, nullptr, nullptr, nullptr, U, DIMD);
  wmma_gemm_t<true, false><<<ggrid, gblk, 0, stream>>>(
      ncr, DIMD, nullptr, r_Wd + DD, DIMD, DIMD,
      nullptr, 0, nullptr, 0, 0, nullptr, nullptr, nullptr, V, DIMD);
  // 7) w = mem_proj * V ; const_b
  wconst_kernel<<<BATCH, 256, 0, stream>>>(memproj, V, ncr, r_bk, r_bd, r_br, wv, constv);
  // 8) t = U + w @ r_Wk^T   (transB, accumulate Cin=U)
  wmma_gemm_t<true, false><<<ggrid, gblk, 0, stream>>>(
      wv, DIMD, nullptr, r_Wk, DIMD, DIMD,
      nullptr, 0, nullptr, 0, 0, nullptr, nullptr, U, tvec, DIMD);
  // 9-11) knowledge streaming: logits -> softmax -> read
  retrieve_kernel<<<dim3(BATCH, KNOW / 64), 256, 0, stream>>>(knowledge, tvec, constv, logits);
  softmax512_kernel<<<BATCH, 256, 0, stream>>>(logits, rattn);
  read_kernel<<<BATCH, 256, 0, stream>>>(knowledge, rattn, readv);
  // 12) m1 = [prev_memory, read] @ w_Wm1 + w_bm1
  wmma_gemm_t<false, false><<<ggrid, gblk, 0, stream>>>(
      h_memories, HLD, nullptr, w_Wm1, DIMD, DIMD,
      readv, DIMD, w_Wm1 + DD, DIMD, DIMD, w_bm1, nullptr, nullptr, m1, DIMD);
  // 13) step attention -> msa, gate
  writeattn_kernel<<<BATCH, 256, 0, stream>>>(h_controls, h_memories, nc, w_Wca,
                                              w_bca, w_Wm3, w_bm3, msa, gate);
  // 14) mp = m1 @ w_Wm2 + msa @ w_Ws + w_bm2 + w_bs
  wmma_gemm_t<false, false><<<ggrid, gblk, 0, stream>>>(
      m1, DIMD, nullptr, w_Wm2, DIMD, DIMD,
      msa, DIMD, w_Ws, DIMD, DIMD, w_bm2, w_bs, nullptr, mpv, DIMD);
  // 15) gated memory update + output assembly
  long tot = (long)BATCH * NSTEP * DIMD;
  output_kernel<<<(int)((tot + 255) / 256), 256, 0, stream>>>(nc, mpv, gate,
                                                              h_controls, h_memories, out);
}
